// MultiHeadLocalAttention2d_46351287058962
// MI455X (gfx1250) — matrix operand order repaired and checked by reading, not compiled, not run on hardware
//
#include <hip/hip_runtime.h>
#include <hip/hip_bf16.h>

// MultiHeadLocalAttention2d for MI455X (gfx1250), wave32 + WMMA bf16 + async LDS.
// B=2, C=OUT_CH=256, H=W=64 (L=4096), HEADS=8 (d=32), KERNEL=7 (K2=49), PAD=3.

typedef __attribute__((ext_vector_type(16))) __bf16 v16bf;
typedef __attribute__((ext_vector_type(8)))  float  v8f;
typedef int v4i_vs __attribute__((vector_size(16)));   // matches builtin param type

union BF16x16 { unsigned short u[16]; v16bf v; };

__device__ __forceinline__ unsigned short f2bf(float f) {
  unsigned int x = __float_as_uint(f);
  x += 0x7FFFu + ((x >> 16) & 1u);        // round-to-nearest-even
  return (unsigned short)(x >> 16);
}
__device__ __forceinline__ unsigned short ld_bf(const float* p)          { return f2bf(*p); }
__device__ __forceinline__ unsigned short ld_bf(const unsigned short* p) { return *p; }

// ---- CDNA5 async global->LDS copy (ASYNCcnt-tracked), guarded -------------
#if defined(__has_builtin)
#if __has_builtin(__builtin_amdgcn_global_load_async_to_lds_b128) && \
    __has_builtin(__builtin_amdgcn_s_wait_asynccnt)
#define HAVE_ASYNC_LDS 1
#endif
#endif
#ifndef HAVE_ASYNC_LDS
#define HAVE_ASYNC_LDS 0
#endif

__device__ __forceinline__ void async_copy16(const void* g, void* l) {
#if HAVE_ASYNC_LDS
  __builtin_amdgcn_global_load_async_to_lds_b128((v4i_vs*)g, (v4i_vs*)l, 0, 0);
#else
  *(uint4*)l = *(const uint4*)g;
#endif
}
__device__ __forceinline__ void async_wait_all() {
#if HAVE_ASYNC_LDS
  __builtin_amdgcn_s_wait_asynccnt(0);
#endif
}

// ---------------------------------------------------------------------------
// Stage 1/3: GEMM  out[o,n] = sum_c W[o,c] * X[c,n] + bias[o]
// M=256, K=256, N=8192. 8 waves/WG -> 32(o) x 64(l) tile, K stepped by 32.
// ---------------------------------------------------------------------------
template <typename TIN, bool OUT_F32>
__global__ void __launch_bounds__(256)
gemm256_kernel(const float* __restrict__ W, const float* __restrict__ bias,
               const TIN* __restrict__ X, void* __restrict__ OUT) {
  const int lane = threadIdx.x & 31;
  const int wid  = threadIdx.x >> 5;
  const int o0   = blockIdx.x * 32 + (wid & 1) * 16;
  const int gn0  = blockIdx.y * 64 + (wid >> 1) * 16;
  const int b    = gn0 >> 12;
  const int l0   = gn0 & 4095;
  const int r    = lane & 15;
  const int half = lane >> 4;

  const int l     = l0 + r;
  const int xbase = b * (256 * 4096) + l;

  v8f acc = {};
  for (int kk = 0; kk < 256; kk += 32) {
    // A operand: weight row o0+r; two 8-float runs per ISA K-half layout.
    BF16x16 a;
    {
      const float* wrow = W + (o0 + r) * 256 + kk + half * 8;
      const float4 w0 = *(const float4*)(wrow);
      const float4 w1 = *(const float4*)(wrow + 4);
      const float4 w2 = *(const float4*)(wrow + 16);
      const float4 w3 = *(const float4*)(wrow + 20);
      a.u[0]  = f2bf(w0.x); a.u[1]  = f2bf(w0.y); a.u[2]  = f2bf(w0.z); a.u[3]  = f2bf(w0.w);
      a.u[4]  = f2bf(w1.x); a.u[5]  = f2bf(w1.y); a.u[6]  = f2bf(w1.z); a.u[7]  = f2bf(w1.w);
      a.u[8]  = f2bf(w2.x); a.u[9]  = f2bf(w2.y); a.u[10] = f2bf(w2.z); a.u[11] = f2bf(w2.w);
      a.u[12] = f2bf(w3.x); a.u[13] = f2bf(w3.y); a.u[14] = f2bf(w3.z); a.u[15] = f2bf(w3.w);
    }
    // B operand: X[c][l], c = kk + half*8 + i (+8 for i>=8) (lane column fixed; coalesced across lanes)
    BF16x16 bx;
    {
      const int cbase = kk + half * 8;
#pragma unroll
      for (int i = 0; i < 16; ++i)
        bx.u[i] = ld_bf(X + xbase + (cbase + i + (i >= 8 ? 8 : 0)) * 4096);
      if (kk + 32 < 256)  // gfx1250 global_prefetch of next K-step tile
        __builtin_prefetch(X + xbase + (cbase + 32) * 4096, 0, 0);
    }
    acc = __builtin_amdgcn_wmma_f32_16x16x32_bf16(false, a.v, false, bx.v,
                                                  (short)0, acc, false, false);
  }

  const int mbase = half * 8;
#pragma unroll
  for (int rr = 0; rr < 8; ++rr) {
    const int o   = o0 + mbase + rr;
    const float v = acc[rr] + bias[o];
    const int idx = b * (256 * 4096) + o * 4096 + l;
    if constexpr (OUT_F32) ((float*)OUT)[idx] = v;
    else                   ((unsigned short*)OUT)[idx] = f2bf(v);
  }
}

// ---------------------------------------------------------------------------
// Stage 2: local attention.
// Block = 64 threads (2 waves); each wave owns a 16-query strip; the two strips
// share one staged 7-row x 48-col halo region (channel-major planes in LDS).
// Region layout: elem = d*PS + r7*RS + cc, cc = kx - x00 + 8, valid kx chunks
// copied 16B-aligned; everything else pre-zeroed (matches zero-pad unfold).
// ---------------------------------------------------------------------------
#define NK 160      // padded key count (154 real: 7 x 22)
#define RS 48       // region row stride (elems)
#define PS 336      // region plane stride per channel d (7*48, 16B-aligned *2B)
#define REGN (32 * PS)

__device__ __forceinline__ void stage_region(const unsigned short* __restrict__ src,
                                             unsigned short* __restrict__ reg,
                                             int y, int x00, int tid) {
  // 32 d-planes x 7 rows x 6 chunks of 8 cols; chunk validity is all-or-nothing.
  for (int cid = tid; cid < 32 * 7 * 6; cid += 64) {
    const int d   = cid / 42;
    const int rm  = cid % 42;
    const int r7  = rm / 6;
    const int c6  = rm % 6;
    const int ky  = y + r7 - 3;
    const int kx0 = x00 - 8 + c6 * 8;
    if ((unsigned)ky < 64u && (unsigned)kx0 <= 56u)
      async_copy16(src + d * 4096 + ky * 64 + kx0,
                   reg + d * PS + r7 * RS + c6 * 8);
  }
}

__global__ void __launch_bounds__(64)
local_attn_kernel(const unsigned short* __restrict__ qb,
                  const unsigned short* __restrict__ kb,
                  const unsigned short* __restrict__ vb,
                  unsigned short* __restrict__ ob) {
  __shared__ unsigned short kvreg[REGN];        // K region, then reused for V
  __shared__ float          sc[2 * 16 * NK];    // fp32 scores per wave
  __shared__ unsigned short at[2 * 16 * NK];    // bf16 attn weights per wave

  const int tid  = threadIdx.x;
  const int lane = tid & 31;
  const int wv   = tid >> 5;
  const int x00  = blockIdx.x * 32;             // strip pair base
  const int y    = blockIdx.y;
  const int b    = blockIdx.z >> 3;
  const int h    = blockIdx.z & 7;
  const int x0   = x00 + wv * 16;
  const int r    = lane & 15;
  const int half = lane >> 4;
  const int chb  = (b * 256 + h * 32) * 4096;

  float*          scw = sc + wv * 16 * NK;
  unsigned short* atw = at + wv * 16 * NK;

  uint4 z4; z4.x = z4.y = z4.z = z4.w = 0;

  // ---- zero + async-stage K region ----
  for (int i = tid; i < REGN / 8; i += 64) ((uint4*)kvreg)[i] = z4;
  __syncthreads();
  stage_region(kb + chb, kvreg, y, x00, tid);
  async_wait_all();
  __syncthreads();

  // ---- Q operand (A): row = query r of this strip ----
  BF16x16 qa;
  {
    const int l  = y * 64 + x0 + r;
    const int kh = half * 8;
#pragma unroll
    for (int i = 0; i < 8; ++i) {
      qa.u[i]     = qb[chb + (kh + i) * 4096 + l];
      qa.u[i + 8] = qb[chb + (16 + kh + i) * 4096 + l];
    }
  }

  // ---- Scores: S[16 x 160] = Q (16x32) * K_region^T, 10 WMMAs ----
#pragma unroll
  for (int t = 0; t < 10; ++t) {
    const int n = t * 16 + r;                       // key column for this lane
    int addr = 0;                                   // pad keys -> addr 0; their scores are unused
    if (n < 154) addr = (n / 22) * RS + (wv * 16 + (n % 22) + 5);
    BF16x16 kv;
    const int db = half * 8;
#pragma unroll
    for (int i = 0; i < 16; ++i)
      kv.u[i] = kvreg[(db + i + (i >= 8 ? 8 : 0)) * PS + addr];
    v8f s = {};
    s = __builtin_amdgcn_wmma_f32_16x16x32_bf16(false, qa.v, false, kv.v,
                                                (short)0, s, false, false);
#pragma unroll
    for (int rr = 0; rr < 8; ++rr)
      scw[(half * 8 + rr) * NK + t * 16 + r] = s[rr];
  }
  __syncthreads();   // scores visible; all waves done reading K region

  // ---- zero attn matrix + region, then overlap V staging with softmax ----
  for (int i = tid; i < REGN / 8; i += 64) ((uint4*)kvreg)[i] = z4;
  for (int i = tid; i < 2 * 16 * NK / 8; i += 64) ((uint4*)at)[i] = z4;
  __syncthreads();
  stage_region(vb + chb, kvreg, y, x00, tid);       // async, in flight

  if (lane < 16) {                                  // softmax while V streams in
    const int m = lane;
    const float scale = 0.0625f;                    // 1/sqrt(256)
    float mx = -3.4e38f;
#pragma unroll
    for (int j = 0; j < 49; ++j) {
      const int n = (j / 7) * 22 + (j % 7) + m;
      mx = fmaxf(mx, scw[m * NK + n] * scale);
    }
    float sum = 0.f;
#pragma unroll
    for (int j = 0; j < 49; ++j) {
      const int n = (j / 7) * 22 + (j % 7) + m;
      sum += __expf(scw[m * NK + n] * scale - mx);
    }
    const float inv = __frcp_rn(sum);
#pragma unroll
    for (int j = 0; j < 49; ++j) {
      const int n = (j / 7) * 22 + (j % 7) + m;
      atw[m * NK + n] = f2bf(__expf(scw[m * NK + n] * scale - mx) * inv);
    }
  }
  async_wait_all();
  __syncthreads();

  // ---- O^T[16 x 32] = attn (16x160) * V_region^T: 2 N-tiles x 5 K-chunks ----
  v8f oacc[2] = {};
#pragma unroll
  for (int c5 = 0; c5 < 5; ++c5) {
    BF16x16 aa;
    {
      const int kh = half * 8;
#pragma unroll
      for (int i = 0; i < 8; ++i) {
        aa.u[i]     = atw[r * NK + c5 * 32 + kh + i];
        aa.u[i + 8] = atw[r * NK + c5 * 32 + 16 + kh + i];
      }
    }
#pragma unroll
    for (int t = 0; t < 2; ++t) {
      const int dcol = t * 16 + r;
      BF16x16 vv;
#pragma unroll
      for (int i = 0; i < 16; ++i) {
        const int key = c5 * 32 + half * 8 + i + (i >= 8 ? 8 : 0);
        int addr = 0;
        if (key < 154) addr = (key / 22) * RS + (wv * 16 + (key % 22) + 5);
        vv.u[i] = kvreg[dcol * PS + addr];
      }
      oacc[t] = __builtin_amdgcn_wmma_f32_16x16x32_bf16(false, aa.v, false, vv.v,
                                                        (short)0, oacc[t], false, false);
    }
  }

#pragma unroll
  for (int t = 0; t < 2; ++t)
#pragma unroll
    for (int rr = 0; rr < 8; ++rr)
      ob[chb + (t * 16 + r) * 4096 + (y * 64 + x0 + half * 8 + rr)] = f2bf(oacc[t][rr]);
}

// ---------------------------------------------------------------------------
extern "C" void kernel_launch(void* const* d_in, const int* in_sizes, int n_in,
                              void* d_out, int out_size, void* d_ws, size_t ws_size,
                              hipStream_t stream) {
  const float* queries = (const float*)d_in[0];
  const float* keys    = (const float*)d_in[1];
  const float* values  = (const float*)d_in[2];
  const float* wq = (const float*)d_in[3];
  const float* bq = (const float*)d_in[4];
  const float* wk = (const float*)d_in[5];
  const float* bk = (const float*)d_in[6];
  const float* wv = (const float*)d_in[7];
  const float* bv = (const float*)d_in[8];
  const float* wo = (const float*)d_in[9];
  const float* bo = (const float*)d_in[10];

  // Workspace: four bf16 tensors of (2, 256, 64, 64) = 2,097,152 elems (16 MB total).
  const size_t NEL = (size_t)2 * 256 * 4096;
  unsigned short* qbuf = (unsigned short*)d_ws;
  unsigned short* kbuf = qbuf + NEL;
  unsigned short* vbuf = kbuf + NEL;
  unsigned short* obuf = vbuf + NEL;

  dim3 ggrid(8, 128), gblk(256);
  gemm256_kernel<float, false><<<ggrid, gblk, 0, stream>>>(wq, bq, queries, (void*)qbuf);
  gemm256_kernel<float, false><<<ggrid, gblk, 0, stream>>>(wk, bk, keys,    (void*)kbuf);
  gemm256_kernel<float, false><<<ggrid, gblk, 0, stream>>>(wv, bv, values,  (void*)vbuf);

  local_attn_kernel<<<dim3(2, 64, 16), dim3(64), 0, stream>>>(qbuf, kbuf, vbuf, obuf);

  gemm256_kernel<unsigned short, true><<<ggrid, gblk, 0, stream>>>(wo, bo, obuf, d_out);
}
